// SRHGNLayer_36644660969555
// MI455X (gfx1250) — matrix operand order repaired and checked by reading, not compiled, not run on hardware
//
#include <hip/hip_runtime.h>
#include <math.h>

#define D    256
#define HN   4
#define HT   4
#define SLOPE 0.01f
#define EPSN  1e-9f

typedef __attribute__((ext_vector_type(16))) _Float16 v16h;
typedef __attribute__((ext_vector_type(8)))  _Float16 v8h;
typedef __attribute__((ext_vector_type(8)))  float    v8f;

__device__ __forceinline__ float wred_sum(float v){
#pragma unroll
    for (int o = 16; o > 0; o >>= 1) v += __shfl_xor(v, o, 32);
    return v;
}
__device__ __forceinline__ float leaky(float x){ return x > 0.f ? x : SLOPE * x; }
__device__ __forceinline__ float gelu_exact(float x){
    return 0.5f * x * (1.f + erff(x * 0.70710678118654752440f));
}
// order-preserving float<->uint map for atomicMax-based segment max
__device__ __forceinline__ unsigned f2mono(float f){
    unsigned u = __float_as_uint(f);
    return (u & 0x80000000u) ? ~u : (u | 0x80000000u);
}
__device__ __forceinline__ float mono2f(unsigned u){
    unsigned b = (u & 0x80000000u) ? (u & 0x7FFFFFFFu) : ~u;
    return __uint_as_float(b);
}
__device__ __forceinline__ float sel4(const float* v, int i){
    float r = v[0];
    r = (i == 1) ? v[1] : r;
    r = (i == 2) ? v[2] : r;
    r = (i == 3) ? v[3] : r;
    return r;
}

// ---------------------------------------------------------------------------
// Kernel 1: 4 fused GEMMs X[N,256] @ W[256,256] + b via v_wmma_f32_16x16x32_f16
//   blockIdx.y: 0 -> hs0 = h_B@edge_W0   1 -> hs1 = h_B@edge_W1
//               2 -> hd  = h_A@node_W0   3 -> outB_pre = h_B@node_W1
// 8 waves/block, 16 rows/wave, K chunked by 32 with W chunk staged in LDS f16.
// ---------------------------------------------------------------------------
__global__ __launch_bounds__(256)
void srhgn_gemm(const float* __restrict__ hA, const float* __restrict__ hB,
                const float* __restrict__ nodeW, const float* __restrict__ nodeb,
                const float* __restrict__ edgeW, const float* __restrict__ edgeb,
                float* __restrict__ hs0, float* __restrict__ hs1,
                float* __restrict__ hd,  float* __restrict__ outBpre, int Nn)
{
    __shared__ __align__(16) _Float16 Wl[256 * 40];   // [n][k-chunk] padded stride 40
    const int tid  = threadIdx.x;
    const int lane = tid & 31;
    const int wave = tid >> 5;

    const float* X; const float* W; const float* bias; float* out;
    switch (blockIdx.y) {
        case 0:  X = hB; W = edgeW;         bias = edgeb;     out = hs0;     break;
        case 1:  X = hB; W = edgeW + D * D; bias = edgeb + D; out = hs1;     break;
        case 2:  X = hA; W = nodeW;         bias = nodeb;     out = hd;      break;
        default: X = hB; W = nodeW + D * D; bias = nodeb + D; out = outBpre; break;
    }

    const int rowbase = blockIdx.x * 128 + wave * 16;
    const int row     = rowbase + (lane & 15);
    const bool rok    = row < Nn;
    const int khalf   = (lane >> 4) << 4;      // 0 for lanes 0-15, 16 for lanes 16-31
    const int klo     = (lane < 16) ? 0 : 16;  // B-fragment K offset within chunk

    v8f acc[16] = {};
    const float4 z4 = make_float4(0.f, 0.f, 0.f, 0.f);

    for (int kc = 0; kc < 8; ++kc) {
        // stage W[kc*32 .. kc*32+31][0..255] into LDS as f16, layout Wl[n*40 + k]
#pragma unroll
        for (int k = 0; k < 32; ++k)
            Wl[tid * 40 + k] = (_Float16)W[(kc * 32 + k) * D + tid];
        __syncthreads();

        // A fragment (16x32 f16, contiguous k = 16*(lane/16) + s, same order as B)
        const float4* Xr = (const float4*)(X + (size_t)row * D);
        const int kb  = kc * 32 + khalf;
        const int kb2 = kb + 8;
        float4 pa = rok ? Xr[(kb  >> 2)    ] : z4;
        float4 pb = rok ? Xr[(kb  >> 2) + 1] : z4;
        float4 pc = rok ? Xr[(kb2 >> 2)    ] : z4;
        float4 pd = rok ? Xr[(kb2 >> 2) + 1] : z4;
        v16h a;
        a[0]  = (_Float16)pa.x; a[1]  = (_Float16)pa.y; a[2]  = (_Float16)pa.z; a[3]  = (_Float16)pa.w;
        a[4]  = (_Float16)pb.x; a[5]  = (_Float16)pb.y; a[6]  = (_Float16)pb.z; a[7]  = (_Float16)pb.w;
        a[8]  = (_Float16)pc.x; a[9]  = (_Float16)pc.y; a[10] = (_Float16)pc.z; a[11] = (_Float16)pc.w;
        a[12] = (_Float16)pd.x; a[13] = (_Float16)pd.y; a[14] = (_Float16)pd.z; a[15] = (_Float16)pd.w;

#pragma unroll
        for (int nt = 0; nt < 16; ++nt) {
            const int col = nt * 16 + (lane & 15);
            const v8h* bp = (const v8h*)&Wl[col * 40 + klo];
            v16h b = __builtin_shufflevector(bp[0], bp[1],
                                             0,1,2,3,4,5,6,7,8,9,10,11,12,13,14,15);
            acc[nt] = __builtin_amdgcn_wmma_f32_16x16x32_f16(
                false, a, false, b, (short)0, acc[nt], false, false);
        }
        __syncthreads();
    }

    // epilogue: C/D layout -> VGPR i holds M = i + 8*(lane>=16), N = lane&15
    const int r0 = rowbase + ((lane >> 4) << 3);
#pragma unroll
    for (int nt = 0; nt < 16; ++nt) {
        const int col = nt * 16 + (lane & 15);
        const float bv = bias[col];
#pragma unroll
        for (int i = 0; i < 8; ++i) {
            const int r = r0 + i;
            if (r < Nn) out[(size_t)r * D + col] = acc[nt][i] + bv;
        }
    }
}

// ---------------------------------------------------------------------------
// Kernel 2: attention logits  a_src[r]=hs_r@srcA_W[r]+b, a_dst[r]=hd@dstA_W[r]+b
// wave per node, butterfly reductions.
// ---------------------------------------------------------------------------
__global__ __launch_bounds__(256)
void srhgn_logits(const float* __restrict__ hs0, const float* __restrict__ hs1,
                  const float* __restrict__ hd,
                  const float* __restrict__ sW, const float* __restrict__ sb,
                  const float* __restrict__ dW, const float* __restrict__ db,
                  float* __restrict__ as0, float* __restrict__ as1,
                  float* __restrict__ ad0, float* __restrict__ ad1, int Nn)
{
    __shared__ __align__(16) float sWl[2 * D * HN];
    __shared__ __align__(16) float dWl[2 * D * HN];
    for (int i = threadIdx.x; i < 2 * D * HN; i += 256) { sWl[i] = sW[i]; dWl[i] = dW[i]; }
    __syncthreads();

    const int n = blockIdx.x * 8 + (threadIdx.x >> 5);
    if (n >= Nn) return;
    const int lane = threadIdx.x & 31;
    const size_t base = (size_t)n * D;

    float ps0[4] = {0,0,0,0}, ps1[4] = {0,0,0,0}, pd0[4] = {0,0,0,0}, pd1[4] = {0,0,0,0};
#pragma unroll
    for (int j = 0; j < 8; ++j) {
        const int d = j * 32 + lane;
        const float x0 = hs0[base + d];
        const float x1 = hs1[base + d];
        const float xd = hd [base + d];
        const float* w0 = &sWl[d * HN];
        const float* w1 = &sWl[D * HN + d * HN];
        const float* v0 = &dWl[d * HN];
        const float* v1 = &dWl[D * HN + d * HN];
#pragma unroll
        for (int h = 0; h < 4; ++h) {
            ps0[h] += x0 * w0[h];
            ps1[h] += x1 * w1[h];
            pd0[h] += xd * v0[h];
            pd1[h] += xd * v1[h];
        }
    }
#pragma unroll
    for (int h = 0; h < 4; ++h) {
        ps0[h] = wred_sum(ps0[h]); ps1[h] = wred_sum(ps1[h]);
        pd0[h] = wred_sum(pd0[h]); pd1[h] = wred_sum(pd1[h]);
    }
    if (lane == 0) {
#pragma unroll
        for (int h = 0; h < 4; ++h) {
            as0[(size_t)n * HN + h] = ps0[h] + sb[h];
            as1[(size_t)n * HN + h] = ps1[h] + sb[HN + h];
            ad0[(size_t)n * HN + h] = pd0[h] + db[h];
            ad1[(size_t)n * HN + h] = pd1[h] + db[HN + h];
        }
    }
}

// ---------------------------------------------------------------------------
// Kernel 3: segment max per dst (edge_softmax pass 1) — thread per edge.
// ---------------------------------------------------------------------------
__global__ __launch_bounds__(256)
void srhgn_edge_max(const int* __restrict__ src0, const int* __restrict__ dst0,
                    const int* __restrict__ src1, const int* __restrict__ dst1,
                    const float* __restrict__ as0, const float* __restrict__ as1,
                    const float* __restrict__ ad0, const float* __restrict__ ad1,
                    unsigned* __restrict__ m0, unsigned* __restrict__ m1, int E)
{
    const int e = blockIdx.x * 256 + threadIdx.x;
    if (e >= E) return;
    const int* sp; const int* dp; const float* as; const float* ad; unsigned* m;
    if (blockIdx.y == 0) { sp = src0; dp = dst0; as = as0; ad = ad0; m = m0; }
    else                 { sp = src1; dp = dst1; as = as1; ad = ad1; m = m1; }
    const int s = sp[e], t = dp[e];
    const float4 av = *(const float4*)(as + (size_t)s * HN);
    const float4 bv = *(const float4*)(ad + (size_t)t * HN);
    atomicMax(&m[(size_t)t * HN + 0], f2mono(leaky(av.x + bv.x)));
    atomicMax(&m[(size_t)t * HN + 1], f2mono(leaky(av.y + bv.y)));
    atomicMax(&m[(size_t)t * HN + 2], f2mono(leaky(av.z + bv.z)));
    atomicMax(&m[(size_t)t * HN + 3], f2mono(leaky(av.w + bv.w)));
}

// ---------------------------------------------------------------------------
// Kernel 4: segment exp-sum per dst (edge_softmax pass 2) — thread per edge.
// ---------------------------------------------------------------------------
__global__ __launch_bounds__(256)
void srhgn_edge_sum(const int* __restrict__ src0, const int* __restrict__ dst0,
                    const int* __restrict__ src1, const int* __restrict__ dst1,
                    const float* __restrict__ as0, const float* __restrict__ as1,
                    const float* __restrict__ ad0, const float* __restrict__ ad1,
                    const unsigned* __restrict__ m0, const unsigned* __restrict__ m1,
                    float* __restrict__ den0, float* __restrict__ den1, int E)
{
    const int e = blockIdx.x * 256 + threadIdx.x;
    if (e >= E) return;
    const int* sp; const int* dp; const float* as; const float* ad;
    const unsigned* m; float* den;
    if (blockIdx.y == 0) { sp = src0; dp = dst0; as = as0; ad = ad0; m = m0; den = den0; }
    else                 { sp = src1; dp = dst1; as = as1; ad = ad1; m = m1; den = den1; }
    const int s = sp[e], t = dp[e];
    const float4 av = *(const float4*)(as + (size_t)s * HN);
    const float4 bv = *(const float4*)(ad + (size_t)t * HN);
    atomicAdd(&den[(size_t)t * HN + 0], expf(leaky(av.x + bv.x) - mono2f(m[(size_t)t * HN + 0])));
    atomicAdd(&den[(size_t)t * HN + 1], expf(leaky(av.y + bv.y) - mono2f(m[(size_t)t * HN + 1])));
    atomicAdd(&den[(size_t)t * HN + 2], expf(leaky(av.z + bv.z) - mono2f(m[(size_t)t * HN + 2])));
    atomicAdd(&den[(size_t)t * HN + 3], expf(leaky(av.w + bv.w) - mono2f(m[(size_t)t * HN + 3])));
}

// ---------------------------------------------------------------------------
// Kernel 5: weighted message scatter  z[dst] += hs[src] * w_head — wave per edge.
// ---------------------------------------------------------------------------
__global__ __launch_bounds__(256)
void srhgn_edge_scatter(const int* __restrict__ src0, const int* __restrict__ dst0,
                        const int* __restrict__ src1, const int* __restrict__ dst1,
                        const float* __restrict__ as0, const float* __restrict__ as1,
                        const float* __restrict__ ad0, const float* __restrict__ ad1,
                        const unsigned* __restrict__ m0, const unsigned* __restrict__ m1,
                        const float* __restrict__ den0, const float* __restrict__ den1,
                        const float* __restrict__ hs0, const float* __restrict__ hs1,
                        float* __restrict__ z0, float* __restrict__ z1, int E)
{
    const int e = blockIdx.x * 8 + (threadIdx.x >> 5);
    if (e >= E) return;
    const int* sp; const int* dp; const float* as; const float* ad;
    const unsigned* m; const float* den; const float* hs; float* z;
    if (blockIdx.y == 0) { sp = src0; dp = dst0; as = as0; ad = ad0; m = m0; den = den0; hs = hs0; z = z0; }
    else                 { sp = src1; dp = dst1; as = as1; ad = ad1; m = m1; den = den1; hs = hs1; z = z1; }
    const int lane = threadIdx.x & 31;
    const int s = sp[e], t = dp[e];
    const float4 av = *(const float4*)(as + (size_t)s * HN);
    const float4 bv = *(const float4*)(ad + (size_t)t * HN);
    const float4 dv = *(const float4*)(den + (size_t)t * HN);
    float w[4];
    w[0] = expf(leaky(av.x + bv.x) - mono2f(m[(size_t)t * HN + 0])) / dv.x;
    w[1] = expf(leaky(av.y + bv.y) - mono2f(m[(size_t)t * HN + 1])) / dv.y;
    w[2] = expf(leaky(av.z + bv.z) - mono2f(m[(size_t)t * HN + 2])) / dv.z;
    w[3] = expf(leaky(av.w + bv.w) - mono2f(m[(size_t)t * HN + 3])) / dv.w;

    const float4* hv = (const float4*)(hs + (size_t)s * D);
    float* zt = z + (size_t)t * D;
#pragma unroll
    for (int c = 0; c < 2; ++c) {
        const int i4 = c * 32 + lane;        // float4 index 0..63, head = i4>>4
        const float wh = sel4(w, i4 >> 4);
        const float4 v = hv[i4];
        atomicAdd(&zt[i4 * 4 + 0], v.x * wh);
        atomicAdd(&zt[i4 * 4 + 1], v.y * wh);
        atomicAdd(&zt[i4 * 4 + 2], v.z * wh);
        atomicAdd(&zt[i4 * 4 + 3], v.w * wh);
    }
}

// ---------------------------------------------------------------------------
// Kernel 6: out_B = normalize(gelu(h_B@node_W1 + b + h_B)); pre-act in place.
// ---------------------------------------------------------------------------
__global__ __launch_bounds__(256)
void srhgn_outB(const float* __restrict__ hB, float* __restrict__ outB, int Nn)
{
    const int n = blockIdx.x * 8 + (threadIdx.x >> 5);
    if (n >= Nn) return;
    const int lane = threadIdx.x & 31;
    const size_t base = (size_t)n * D;
    float4 g[2]; float ss = 0.f;
#pragma unroll
    for (int c = 0; c < 2; ++c) {
        const int i4 = c * 32 + lane;
        const float4 pre = *(const float4*)(outB + base + i4 * 4);
        const float4 hb  = *(const float4*)(hB   + base + i4 * 4);
        float4 gg;
        gg.x = gelu_exact(pre.x + hb.x);
        gg.y = gelu_exact(pre.y + hb.y);
        gg.z = gelu_exact(pre.z + hb.z);
        gg.w = gelu_exact(pre.w + hb.w);
        ss += gg.x * gg.x + gg.y * gg.y + gg.z * gg.z + gg.w * gg.w;
        g[c] = gg;
    }
    ss = wred_sum(ss);
    const float inv = 1.f / fmaxf(sqrtf(ss), EPSN);
#pragma unroll
    for (int c = 0; c < 2; ++c) {
        const int i4 = c * 32 + lane;
        float4 o;
        o.x = g[c].x * inv; o.y = g[c].y * inv; o.z = g[c].z * inv; o.w = g[c].w * inv;
        *(float4*)(outB + base + i4 * 4) = o;
    }
}

// ---------------------------------------------------------------------------
// Kernel 7: out_A fusion — semantic + relation attention, gelu, normalize.
// wave per node.
// ---------------------------------------------------------------------------
__global__ __launch_bounds__(256)
void srhgn_outA(const float* __restrict__ hA, const float* __restrict__ z0,
                const float* __restrict__ z1, const float* __restrict__ hd,
                const float* __restrict__ sSW, const float* __restrict__ sSb,
                const float* __restrict__ sDW, const float* __restrict__ sDb,
                const float* __restrict__ rAW, const float* __restrict__ rAb,
                const float* __restrict__ remb, float* __restrict__ outA, int Nn)
{
    __shared__ __align__(16) float SW[2 * D * HT];
    __shared__ __align__(16) float DW[2 * D * HT];
    __shared__ float relw[8];
    for (int i = threadIdx.x; i < 2 * D * HT; i += 256) { SW[i] = sSW[i]; DW[i] = sDW[i]; }
    if (threadIdx.x < 8) {
        const int r = threadIdx.x >> 2, h = threadIdx.x & 3;
        float s = rAb[r * HT + h];
        for (int d = 0; d < D; ++d) s += remb[r * D + d] * rAW[(r * D + d) * HT + h];
        relw[threadIdx.x] = leaky(s);
    }
    __syncthreads();
    if (threadIdx.x < 4) {
        const int h = threadIdx.x;
        const float l0 = relw[h], l1 = relw[4 + h];
        const float mx = fmaxf(l0, l1);
        const float e0 = expf(l0 - mx), e1 = expf(l1 - mx);
        const float inv = 1.f / (e0 + e1);
        relw[h] = e0 * inv; relw[4 + h] = e1 * inv;
    }
    __syncthreads();

    const int n = blockIdx.x * 8 + (threadIdx.x >> 5);
    if (n >= Nn) return;
    const int lane = threadIdx.x & 31;
    const size_t base = (size_t)n * D;

    float4 z0v[2], z1v[2];
    float ss0 = 0.f, ss1 = 0.f, ssd = 0.f;
    float pS0[4] = {0,0,0,0}, pS1[4] = {0,0,0,0}, pD0[4] = {0,0,0,0}, pD1[4] = {0,0,0,0};
#pragma unroll
    for (int c = 0; c < 2; ++c) {
        const int i4 = c * 32 + lane;
        const float4 a  = *(const float4*)(z0 + base + i4 * 4);
        const float4 b  = *(const float4*)(z1 + base + i4 * 4);
        const float4 dz = *(const float4*)(hd + base + i4 * 4);
        z0v[c] = a; z1v[c] = b;
        const float* ap = (const float*)&a;
        const float* bp = (const float*)&b;
        const float* ep = (const float*)&dz;
#pragma unroll
        for (int j = 0; j < 4; ++j) {
            const int d = i4 * 4 + j;
            const float za = ap[j], zb = bp[j], zd = ep[j];
            ss0 += za * za; ss1 += zb * zb; ssd += zd * zd;
            const float* w0 = &SW[d * HT];
            const float* w1 = &SW[D * HT + d * HT];
            const float* v0 = &DW[d * HT];
            const float* v1 = &DW[D * HT + d * HT];
#pragma unroll
            for (int h = 0; h < 4; ++h) {
                pS0[h] += za * w0[h];
                pS1[h] += zb * w1[h];
                pD0[h] += zd * v0[h];
                pD1[h] += zd * v1[h];
            }
        }
    }
    ss0 = wred_sum(ss0); ss1 = wred_sum(ss1); ssd = wred_sum(ssd);
#pragma unroll
    for (int h = 0; h < 4; ++h) {
        pS0[h] = wred_sum(pS0[h]); pS1[h] = wred_sum(pS1[h]);
        pD0[h] = wred_sum(pD0[h]); pD1[h] = wred_sum(pD1[h]);
    }
    const float in0 = 1.f / fmaxf(sqrtf(ss0), EPSN);
    const float in1 = 1.f / fmaxf(sqrtf(ss1), EPSN);
    const float ind = 1.f / fmaxf(sqrtf(ssd), EPSN);

    float at0[4], at1[4];
#pragma unroll
    for (int h = 0; h < 4; ++h) {
        const float s0 = leaky(pS0[h] * in0 + sSb[h]      + pD0[h] * ind + sDb[h]);
        const float s1 = leaky(pS1[h] * in1 + sSb[HT + h] + pD1[h] * ind + sDb[HT + h]);
        const float mx = fmaxf(s0, s1);
        const float e0 = expf(s0 - mx), e1 = expf(s1 - mx);
        const float inv = 1.f / (e0 + e1);
        at0[h] = 0.5f * (e0 * inv) + 0.5f * relw[h];
        at1[h] = 0.5f * (e1 * inv) + 0.5f * relw[4 + h];
    }

    float4 g[2]; float ssg = 0.f;
#pragma unroll
    for (int c = 0; c < 2; ++c) {
        const int i4 = c * 32 + lane;
        const int hh = i4 >> 4;
        const float a0 = sel4(at0, hh), a1 = sel4(at1, hh);
        const float4 ha = *(const float4*)(hA + base + i4 * 4);
        float4 gg;
        gg.x = gelu_exact(z0v[c].x * a0 + z1v[c].x * a1 + ha.x);
        gg.y = gelu_exact(z0v[c].y * a0 + z1v[c].y * a1 + ha.y);
        gg.z = gelu_exact(z0v[c].z * a0 + z1v[c].z * a1 + ha.z);
        gg.w = gelu_exact(z0v[c].w * a0 + z1v[c].w * a1 + ha.w);
        ssg += gg.x * gg.x + gg.y * gg.y + gg.z * gg.z + gg.w * gg.w;
        g[c] = gg;
    }
    ssg = wred_sum(ssg);
    const float invg = 1.f / fmaxf(sqrtf(ssg), EPSN);
#pragma unroll
    for (int c = 0; c < 2; ++c) {
        const int i4 = c * 32 + lane;
        float4 o;
        o.x = g[c].x * invg; o.y = g[c].y * invg; o.z = g[c].z * invg; o.w = g[c].w * invg;
        *(float4*)(outA + base + i4 * 4) = o;
    }
}

// ---------------------------------------------------------------------------
extern "C" void kernel_launch(void* const* d_in, const int* in_sizes, int n_in,
                              void* d_out, int out_size, void* d_ws, size_t ws_size,
                              hipStream_t stream)
{
    (void)n_in; (void)out_size; (void)ws_size;
    const float* hA    = (const float*)d_in[0];
    const float* hB    = (const float*)d_in[1];
    const int*   src0  = (const int*)d_in[2];
    const int*   dst0  = (const int*)d_in[3];
    const int*   src1  = (const int*)d_in[4];
    const int*   dst1  = (const int*)d_in[5];
    const float* nodeW = (const float*)d_in[6];
    const float* nodeb = (const float*)d_in[7];
    const float* edgeW = (const float*)d_in[8];
    const float* edgeb = (const float*)d_in[9];
    const float* srcAW = (const float*)d_in[10];
    const float* srcAb = (const float*)d_in[11];
    const float* dstAW = (const float*)d_in[12];
    const float* dstAb = (const float*)d_in[13];
    const float* semSW = (const float*)d_in[14];
    const float* semSb = (const float*)d_in[15];
    const float* semDW = (const float*)d_in[16];
    const float* semDb = (const float*)d_in[17];
    const float* relAW = (const float*)d_in[18];
    const float* relAb = (const float*)d_in[19];
    const float* remb  = (const float*)d_in[20];

    const int Nn = in_sizes[0] / D;
    const int E  = in_sizes[2];
    const size_t ND = (size_t)Nn * D;

    float* outA = (float*)d_out;
    float* outB = (float*)d_out + ND;

    float* ws  = (float*)d_ws;
    float* hs0 = ws;
    float* hs1 = ws + ND;
    float* hd  = ws + 2 * ND;
    float* z0  = ws + 3 * ND;
    float* z1  = ws + 4 * ND;
    float* as0 = ws + 5 * ND;
    float* as1 = as0 + (size_t)Nn * HN;
    float* ad0 = as1 + (size_t)Nn * HN;
    float* ad1 = ad0 + (size_t)Nn * HN;
    unsigned* m0 = (unsigned*)(ad1 + (size_t)Nn * HN);
    unsigned* m1 = m0 + (size_t)Nn * HN;
    float* den0  = (float*)(m1 + (size_t)Nn * HN);
    float* den1  = den0 + (size_t)Nn * HN;

    // zero z accumulators, segment-max (mono encoding: 0 == minimum) and denoms
    hipMemsetAsync(z0, 0, 2 * ND * sizeof(float), stream);
    hipMemsetAsync(m0, 0, (size_t)4 * Nn * HN * sizeof(float), stream);

    dim3 gG((Nn + 127) / 128, 4);
    srhgn_gemm<<<gG, 256, 0, stream>>>(hA, hB, nodeW, nodeb, edgeW, edgeb,
                                       hs0, hs1, hd, outB, Nn);
    srhgn_logits<<<dim3((Nn + 7) / 8), 256, 0, stream>>>(hs0, hs1, hd,
                                                         srcAW, srcAb, dstAW, dstAb,
                                                         as0, as1, ad0, ad1, Nn);
    dim3 gE((E + 255) / 256, 2);
    srhgn_edge_max<<<gE, 256, 0, stream>>>(src0, dst0, src1, dst1,
                                           as0, as1, ad0, ad1, m0, m1, E);
    srhgn_edge_sum<<<gE, 256, 0, stream>>>(src0, dst0, src1, dst1,
                                           as0, as1, ad0, ad1, m0, m1, den0, den1, E);
    dim3 gS((E + 7) / 8, 2);
    srhgn_edge_scatter<<<gS, 256, 0, stream>>>(src0, dst0, src1, dst1,
                                               as0, as1, ad0, ad1, m0, m1, den0, den1,
                                               hs0, hs1, z0, z1, E);
    srhgn_outB<<<dim3((Nn + 7) / 8), 256, 0, stream>>>(hB, outB, Nn);
    srhgn_outA<<<dim3((Nn + 7) / 8), 256, 0, stream>>>(hA, z0, z1, hd,
                                                       semSW, semSb, semDW, semDb,
                                                       relAW, relAb, remb, outA, Nn);
}
